// Spatial_Attn_15066745274571
// MI455X (gfx1250) — compile-verified, matrix operand order checked by reading, not run on hardware
//
#include <hip/hip_runtime.h>
#include <hip/hip_bf16.h>

typedef __attribute__((ext_vector_type(16))) _Float16 v16h;
typedef __attribute__((ext_vector_type(8)))  float    v8f;

#define BATCH 16
#define CH    64
#define NPIX  1920            // 24*80
#define NT    15              // NPIX / 128
#define TILE_N 128
#define KSTEP 32
#define NKS   (NPIX/KSTEP)    // 60

// ---- workspace layout (bytes) ----
#define WS_PV   0                                   // B*64*N f16
#define WS_XH   (WS_PV + BATCH*CH*NPIX*2)           // B*64*N f16
#define WS_MCA  (WS_XH + BATCH*CH*NPIX*2)           // 64*64 f16
#define WS_MX   (WS_MCA + CH*CH*2)
#define WS_V    (WS_MX + CH*CH*2)                   // 64 f32

// ---- LDS layout for main kernel (bytes) ----
#define SA0 0          // 64*40 f16  = 5120
#define SB0 5120       // 128*40 f16 = 10240
#define SBUFD 15360    // SA1 = SA0+15360, SB1 = SB0+15360 (ends 30720)
#define SCX 0          // 128*136 f16 = 34816 (aliases staging; used after K loop)
#define SMM 34816      // 64*136 f16  = 17408
#define SNB 52224      // 256 f32
#define SNF 53248      // 128 f32
#define SVV 53760      // 64 f32
#define SMEM_BYTES 54016

union F16x16 { v16h v; uint4 q[2]; };
union H8     { uint4 q; _Float16 h[8]; };

// ---------------------------------------------------------------------------
// prep: fold BN+conv params into M_ca, M_x (f16) and bias v (f32)
// out = M_ca @ ca + M_x @ x + v + x
// ---------------------------------------------------------------------------
__global__ __launch_bounds__(64) void sa_prep_kernel(
    const float* __restrict__ ca_w,  const float* __restrict__ ca_b,
    const float* __restrict__ ca_g,  const float* __restrict__ ca_bb,
    const float* __restrict__ ca_m,  const float* __restrict__ ca_v,
    const float* __restrict__ ctx_w, const float* __restrict__ ctx_b,
    const float* __restrict__ ctx_g, const float* __restrict__ ctx_bb,
    const float* __restrict__ ctx_m, const float* __restrict__ ctx_v,
    const float* __restrict__ w1,    const float* __restrict__ b1,
    const float* __restrict__ g1,    const float* __restrict__ bb1,
    const float* __restrict__ m1,    const float* __restrict__ v1,
    const float* __restrict__ gamma,
    _Float16* __restrict__ Mca, _Float16* __restrict__ Mx, float* __restrict__ vout)
{
  __shared__ float A1[64*64], A2[64*64], be1[64], be2[64];
  const int c = threadIdx.x;
  float inv_ca = ca_g[c]  * rsqrtf(ca_v[c]  + 1e-5f);
  float inv_cx = ctx_g[c] * rsqrtf(ctx_v[c] + 1e-5f);
  for (int k = 0; k < 64; ++k) {
    A1[c*64+k] = inv_ca * ca_w[c*64+k];
    A2[c*64+k] = inv_cx * ctx_w[c*64+k];
  }
  be1[c] = ca_bb[c]  + (ca_b[c]  - ca_m[c])  * inv_ca;
  be2[c] = ctx_bb[c] + (ctx_b[c] - ctx_m[c]) * inv_cx;
  __syncthreads();
  const float g    = gamma[0];
  const float inv1 = g1[c] * rsqrtf(v1[c] + 1e-5f);
  for (int k = 0; k < 64; ++k) {
    float sca = 0.f, sx = 0.f;
    for (int t = 0; t < 64; ++t) {
      sca += w1[c*128 + t]      * A1[t*64 + k];
      sx  += w1[c*128 + 64 + t] * A2[t*64 + k];
    }
    Mca[c*64 + k] = (_Float16)(g * inv1 * sca);
    Mx [c*64 + k] = (_Float16)(g * inv1 * sx);
  }
  float sb = b1[c];
  for (int t = 0; t < 64; ++t)
    sb += w1[c*128 + t] * be1[t] + w1[c*128 + 64 + t] * be2[t];
  vout[c] = g * (inv1 * (sb - m1[c]) + bb1[c]);
}

// ---------------------------------------------------------------------------
// pv: value conv per pixel -> pv f16; also f16 copy of x
// ---------------------------------------------------------------------------
__global__ __launch_bounds__(256) void sa_pv_kernel(
    const float* __restrict__ x, const float* __restrict__ vw,
    const float* __restrict__ vb,
    _Float16* __restrict__ pv, _Float16* __restrict__ xh)
{
  __shared__ float w[64*64];
  __shared__ float bsh[64];
  const int tid = threadIdx.x;
  for (int i = tid; i < 4096; i += 256) w[i] = vw[i];
  if (tid < 64) bsh[tid] = vb[tid];
  __syncthreads();
  const int pix = blockIdx.x * 256 + tid;
  const int b = pix / NPIX, i = pix % NPIX;
  const size_t base = (size_t)b * CH * NPIX + i;
  float xv[64];
  #pragma unroll
  for (int k = 0; k < 64; ++k) {
    xv[k] = x[base + (size_t)k * NPIX];
    xh[base + (size_t)k * NPIX] = (_Float16)xv[k];
  }
  for (int c = 0; c < 64; ++c) {
    float acc = bsh[c];
    #pragma unroll
    for (int k = 0; k < 64; ++k) acc += w[c*64 + k] * xv[k];
    pv[base + (size_t)c * NPIX] = (_Float16)acc;
  }
}

// ---------------------------------------------------------------------------
// main fused kernel: attn=exp(-2d) (written f32 to output), column-sum
// normalizer, ca = pv@attn / norm  (WMMA f16->f32), then
// out = [Mca|Mx] @ [ca;x] + v + x   (second WMMA GEMM) — one block per
// (batch, 128-column slab); double-buffered LDS staging.
// ---------------------------------------------------------------------------
__global__ __launch_bounds__(256) void sa_attn_gemm_kernel(
    const float* __restrict__ x, const float* __restrict__ dist,
    const _Float16* __restrict__ pv, const _Float16* __restrict__ xh,
    const _Float16* __restrict__ Mca, const _Float16* __restrict__ Mx,
    const float* __restrict__ vvec,
    float* __restrict__ out, float* __restrict__ attn)
{
  __shared__ __align__(16) char smem[SMEM_BYTES];
  const int tid  = threadIdx.x;
  const int b    = blockIdx.x / NT;
  const int jt   = blockIdx.x % NT;
  const int j0   = jt * TILE_N;
  const int lane = tid & 31, wv = tid >> 5, lo = lane & 15, kh = lane >> 4;
  const int colT = wv * 16 + lo;                  // column within tile [0,128)
  const int ac   = tid >> 2, akk = (tid & 3) * 8; // A staging ids
  const int cc   = tid & 127, r0 = tid >> 7;      // B staging: col cc, rows r0*16..+15
  const size_t pvb = (size_t)b * CH * NPIX;
  const size_t db  = (size_t)b * NPIX * NPIX;

  v8f acc[4] = {};
  float csum = 0.f;
  uint4 areg;
  float ereg[16];

  auto stage_load = [&](int ks) {
    areg = *(const uint4*)(pv + pvb + (size_t)ac * NPIX + ks + akk);
    const float* dp = dist + db + (size_t)(ks + r0 * 16) * NPIX + j0 + cc;
    float*       ap = attn + db + (size_t)(ks + r0 * 16) * NPIX + j0 + cc;
    #pragma unroll
    for (int it = 0; it < 16; ++it) {
      const float e = __expf(-2.0f * dp[(size_t)it * NPIX]);
      ap[(size_t)it * NPIX] = e;            // attn output (f32)
      csum += e;                             // column-sum normalizer
      ereg[it] = e;
    }
  };
  auto stage_store = [&](int buf) {
    _Float16* sA = (_Float16*)(smem + SA0 + buf * SBUFD);
    _Float16* sB = (_Float16*)(smem + SB0 + buf * SBUFD);
    *(uint4*)(sA + ac * 40 + akk) = areg;
    #pragma unroll
    for (int q = 0; q < 2; ++q) {
      H8 hq;
      #pragma unroll
      for (int e2 = 0; e2 < 8; ++e2) hq.h[e2] = (_Float16)ereg[q * 8 + e2];
      *(uint4*)(sB + cc * 40 + r0 * 16 + q * 8) = hq.q;   // 2x ds_store_b128
    }
  };
  auto compute = [&](int buf) {
    const _Float16* sA = (const _Float16*)(smem + SA0 + buf * SBUFD);
    const _Float16* sB = (const _Float16*)(smem + SB0 + buf * SBUFD);
    // preload ALL fragments, then back-to-back independent WMMAs
    F16x16 bf, af[4];
    bf.q[0] = *(const uint4*)(sB + colT * 40 + kh * 8);
    bf.q[1] = *(const uint4*)(sB + colT * 40 + 16 + kh * 8);
    #pragma unroll
    for (int mt = 0; mt < 4; ++mt) {
      const int row = mt * 16 + lo;
      af[mt].q[0] = *(const uint4*)(sA + row * 40 + kh * 8);
      af[mt].q[1] = *(const uint4*)(sA + row * 40 + 16 + kh * 8);
    }
    #pragma unroll
    for (int mt = 0; mt < 4; ++mt)
      acc[mt] = __builtin_amdgcn_wmma_f32_16x16x32_f16(
          false, af[mt].v, false, bf.v, (short)0, acc[mt], false, false);
  };

  // load epilogue operands into non-aliased LDS regions
  {
    _Float16* sM = (_Float16*)(smem + SMM);
    for (int idx = tid; idx < 64 * 128; idx += 256) {
      const int m = idx >> 7, k = idx & 127;
      sM[m * 136 + k] = (k < 64) ? Mca[m * 64 + k] : Mx[m * 64 + (k - 64)];
    }
    if (tid < 64) ((float*)(smem + SVV))[tid] = vvec[tid];
  }

  stage_load(0);
  stage_store(0);
  __syncthreads();

  for (int s = 0; s < NKS; ++s) {
    const int cur = s & 1;
    if (s + 1 < NKS) stage_load((s + 1) * KSTEP);   // global loads + exp
    compute(cur);                                    // WMMA on current buffer
    if (s + 1 < NKS) stage_store(cur ^ 1);           // fill other buffer
    __syncthreads();
  }

  // ---- normalizer + build 128x128 B operand [ca ; x] in sCX ----
  float* nb = (float*)(smem + SNB);
  nb[tid] = csum;
  _Float16* sC = (_Float16*)(smem + SCX);
  {
    const int ci = tid >> 2;
    const int c0 = (tid & 3) * 32;
    const _Float16* xp = xh + pvb + (size_t)ci * NPIX + j0 + c0;
    #pragma unroll
    for (int q = 0; q < 4; ++q) {
      H8 hq; hq.q = *(const uint4*)(xp + q * 8);
      #pragma unroll
      for (int e2 = 0; e2 < 8; ++e2)
        sC[(c0 + q * 8 + e2) * 136 + 64 + ci] = hq.h[e2];
    }
  }
  __syncthreads();
  float* nf = (float*)(smem + SNF);
  if (tid < 128) nf[tid] = 1.0f / (nb[tid] + nb[tid + 128]);
  __syncthreads();
  {
    const float rn = nf[colT];
    #pragma unroll
    for (int mt = 0; mt < 4; ++mt) {
      H8 hq;
      #pragma unroll
      for (int r = 0; r < 8; ++r) hq.h[r] = (_Float16)(acc[mt][r] * rn);
      *(uint4*)(sC + colT * 136 + mt * 16 + kh * 8) = hq.q;
    }
  }
  __syncthreads();

  // ---- second GEMM: out64x128 = [Mca|Mx](64x128) @ [ca;x](128x128) ----
  v8f o[4] = {};
  const _Float16* sM = (const _Float16*)(smem + SMM);
  #pragma unroll
  for (int ks2 = 0; ks2 < 128; ks2 += 32) {
    F16x16 bf, af[4];
    bf.q[0] = *(const uint4*)(sC + colT * 136 + ks2 + kh * 8);
    bf.q[1] = *(const uint4*)(sC + colT * 136 + ks2 + 16 + kh * 8);
    #pragma unroll
    for (int mt = 0; mt < 4; ++mt) {
      const int row = mt * 16 + lo;
      af[mt].q[0] = *(const uint4*)(sM + row * 136 + ks2 + kh * 8);
      af[mt].q[1] = *(const uint4*)(sM + row * 136 + ks2 + 16 + kh * 8);
    }
    #pragma unroll
    for (int mt = 0; mt < 4; ++mt)
      o[mt] = __builtin_amdgcn_wmma_f32_16x16x32_f16(
          false, af[mt].v, false, bf.v, (short)0, o[mt], false, false);
  }

  // ---- epilogue: + v + residual x, store fp32 ----
  const float* sV = (const float*)(smem + SVV);
  const int j = j0 + colT;
  #pragma unroll
  for (int mt = 0; mt < 4; ++mt) {
    #pragma unroll
    for (int r = 0; r < 8; ++r) {
      const int m = mt * 16 + kh * 8 + r;
      const size_t oi = pvb + (size_t)m * NPIX + j;
      out[oi] = o[mt][r] + sV[m] + x[oi];
    }
  }
}

// ---------------------------------------------------------------------------
extern "C" void kernel_launch(void* const* d_in, const int* in_sizes, int n_in,
                              void* d_out, int out_size, void* d_ws, size_t ws_size,
                              hipStream_t stream) {
  (void)in_sizes; (void)n_in; (void)out_size; (void)ws_size;
  const float* x      = (const float*)d_in[0];
  const float* dist   = (const float*)d_in[1];
  const float* vw     = (const float*)d_in[2];
  const float* vb     = (const float*)d_in[3];
  const float* ca_w   = (const float*)d_in[4];
  const float* ca_b   = (const float*)d_in[5];
  const float* ca_g   = (const float*)d_in[6];
  const float* ca_bb  = (const float*)d_in[7];
  const float* ca_m   = (const float*)d_in[8];
  const float* ca_v   = (const float*)d_in[9];
  const float* ctx_w  = (const float*)d_in[10];
  const float* ctx_b  = (const float*)d_in[11];
  const float* ctx_g  = (const float*)d_in[12];
  const float* ctx_bb = (const float*)d_in[13];
  const float* ctx_m  = (const float*)d_in[14];
  const float* ctx_v  = (const float*)d_in[15];
  const float* w1     = (const float*)d_in[16];
  const float* b1     = (const float*)d_in[17];
  const float* g1     = (const float*)d_in[18];
  const float* bb1    = (const float*)d_in[19];
  const float* m1     = (const float*)d_in[20];
  const float* v1     = (const float*)d_in[21];
  const float* gamma  = (const float*)d_in[22];

  char* ws = (char*)d_ws;
  _Float16* pv  = (_Float16*)(ws + WS_PV);
  _Float16* xh  = (_Float16*)(ws + WS_XH);
  _Float16* Mca = (_Float16*)(ws + WS_MCA);
  _Float16* Mx  = (_Float16*)(ws + WS_MX);
  float*    vv  = (float*)(ws + WS_V);

  float* out  = (float*)d_out;
  float* attn = out + (size_t)BATCH * CH * NPIX;   // tuple: (out, attn)

  sa_prep_kernel<<<1, 64, 0, stream>>>(ca_w, ca_b, ca_g, ca_bb, ca_m, ca_v,
                                       ctx_w, ctx_b, ctx_g, ctx_bb, ctx_m, ctx_v,
                                       w1, b1, g1, bb1, m1, v1, gamma, Mca, Mx, vv);
  sa_pv_kernel<<<(BATCH * NPIX) / 256, 256, 0, stream>>>(x, vw, vb, pv, xh);
  sa_attn_gemm_kernel<<<BATCH * NT, 256, 0, stream>>>(x, dist, pv, xh, Mca, Mx, vv,
                                                      out, attn);
}
